// CausalDit_10428180595096
// MI455X (gfx1250) — compile-verified, matrix operand order checked by reading, not run on hardware
//
#include <hip/hip_runtime.h>
#include <math.h>

// ---------------- model constants ----------------
#define DUR   6
#define CCH   3
#define HH    32
#define WW    32
#define PP    2
#define DD    512
#define NHH   8
#define DHH   64
#define EXPF  4
#define NBLK  6
#define TT    1000
#define GRIDN 256          // (H/P)*(W/P)
#define SEQL  257          // GRID + NREG
#define NZ    (DUR*SEQL)         // 1542
#define NX    ((DUR-1)*SEQL)     // 1285
#define NKV   (NZ+NX)            // 2827

// ---------------- WMMA types ----------------
typedef __attribute__((ext_vector_type(8)))  float  v8f;
typedef __attribute__((ext_vector_type(16))) __bf16 v16bf;
typedef __attribute__((ext_vector_type(8)))  __bf16 v8bf;
typedef __attribute__((ext_vector_type(4)))  unsigned int v4u;
typedef __attribute__((ext_vector_type(8)))  int v8i;
typedef __attribute__((ext_vector_type(4)))  int v4i;

#if __has_builtin(__builtin_amdgcn_tensor_load_to_lds) && __has_builtin(__builtin_amdgcn_s_wait_tensorcnt)
#define HAVE_TDM 1
#endif

#define LDA 40    // bf16 elems per LDS row (80B, 16B-aligned rows)
#define LDB 132   // f32 elems per LDS row for TDM-staged B (128 + 4 pad)

__device__ inline v8f zero_v8f() {
  v8f r;
#pragma unroll
  for (int i = 0; i < 8; i++) r[i] = 0.f;
  return r;
}
__device__ inline v16bf make_frag(v8bf lo, v8bf hi) {
  v16bf a;
#pragma unroll
  for (int i = 0; i < 8; i++) { a[i] = lo[i]; a[8 + i] = hi[i]; }
  return a;
}
// 8 consecutive f32 (32B aligned) -> bf16, branchless zero-select
__device__ inline v8bf cvt8sel(const float* p, bool ok, float s) {
  float4 a = *(const float4*)p;
  float4 b = *(const float4*)(p + 4);
  v8bf r;
  r[0] = (__bf16)(ok ? a.x * s : 0.f); r[1] = (__bf16)(ok ? a.y * s : 0.f);
  r[2] = (__bf16)(ok ? a.z * s : 0.f); r[3] = (__bf16)(ok ? a.w * s : 0.f);
  r[4] = (__bf16)(ok ? b.x * s : 0.f); r[5] = (__bf16)(ok ? b.y * s : 0.f);
  r[6] = (__bf16)(ok ? b.z * s : 0.f); r[7] = (__bf16)(ok ? b.w * s : 0.f);
  return r;
}
__device__ inline float gelu_tanh(float x) {
  float x3 = x * x * x;
  return 0.5f * x * (1.f + tanhf(0.7978845608f * (x + 0.044715f * x3)));
}

// ---- TDM: 2D tile (tile 128 x 32 rows, f32) global -> LDS, padded rows ----
// D# per CDNA5 ISA ch. 8: group0 {count,lds_addr,global_addr,type=2},
// group1 {data_size=4B, pad 4 DWORDs every 128 (row stride 132), dims/strides}.
// amdgpu-toolchain 6-arg builtin: (v4u g0, v8i g1, v4i g2, v4i g3, v8i g4, i32 cpol)
__device__ inline void tdm_load_tile(const float* g, unsigned lds_byte,
                                     unsigned wrem, unsigned hrem, unsigned stride)
{
#ifdef HAVE_TDM
  unsigned long long ga = (unsigned long long)(const void*)g;
  v4u g0;
  g0[0] = 1u;                                    // count=1, gather off
  g0[1] = lds_byte;                              // lds_addr [63:32]
  g0[2] = (unsigned)(ga & 0xFFFFFFFFu);          // global_addr low
  g0[3] = (unsigned)((ga >> 32) & 0x1FFFFFFu) | (2u << 30);  // addr hi | type=2
  v8i g1;
  g1[0] = (int)((2u << 16) | (1u << 20) | (6u << 22) | (3u << 25)); // 4B, pad en, 128 dw interval, 4 dw pad
  g1[1] = (int)((wrem & 0xFFFFu) << 16);                       // tensor_dim0 lo
  g1[2] = (int)(((wrem >> 16) & 0xFFFFu) | ((hrem & 0xFFFFu) << 16)); // dim0 hi | dim1 lo
  g1[3] = (int)(((hrem >> 16) & 0xFFFFu) | (128u << 16));      // dim1 hi | tile_dim0=128
  g1[4] = (int)32u;                                            // tile_dim1=32
  g1[5] = (int)stride;                                         // tensor_dim0_stride lo
  g1[6] = 0;
  g1[7] = 0;
  v4i z4; z4[0] = z4[1] = z4[2] = z4[3] = 0;
  v8i z8;
#pragma unroll
  for (int i = 0; i < 8; i++) z8[i] = 0;
  __builtin_amdgcn_tensor_load_to_lds(g0, g1, z4, z4, z8, 0);
#else
  (void)g; (void)lds_byte; (void)wrem; (void)hrem; (void)stride;
#endif
}

// =====================================================================
// GEMM: Y[M,N] = act(X[M,K] @ W[K,N] + bias[N])
// 256 threads (8 waves). Block tile 128x128; wave tile 32x64 (8 wmma/k-step).
// A: f32 -> bf16 register-pipelined staging. B: TDM async copy (f32,
// double-buffered, s_wait_tensorcnt) with cooperative fallback.
// =====================================================================
#define ABYTES (128 * LDA * 2)         // 10240
#define BBYTES (32 * LDB * 4)          // 16896
#define SMEMB  (ABYTES + 2 * BBYTES)   // 44032

__global__ __launch_bounds__(256) void gemm_bf16(
    const float* __restrict__ X, const float* __restrict__ W,
    const float* __restrict__ bias, float* __restrict__ Y,
    int M, int N, int K, int act)
{
  // single LDS block; assumed to start at LDS offset 0 for the TDM descriptor
  __shared__ __align__(16) char smem[SMEMB];
  __bf16* As  = (__bf16*)smem;
  float*  Bs0 = (float*)(smem + ABYTES);
  float*  Bs1 = (float*)(smem + ABYTES + BBYTES);

  const int tid  = threadIdx.x;
  const int lane = tid & 31;
  const int wave = tid >> 5;
  const int wm = wave & 3, wn = wave >> 2;
  const int bm = blockIdx.x * 128;
  const int bn = blockIdx.y * 128;
  const int hf  = lane >> 4;
  const int l16 = lane & 15;
  const int nt = K / 32;

  v8f acc[2][4];
#pragma unroll
  for (int m = 0; m < 2; m++)
#pragma unroll
    for (int j = 0; j < 4; j++) acc[m][j] = zero_v8f();

  float2 av[8];
  auto loadA = [&](int it) {
    int kt = it * 32;
#pragma unroll
    for (int i = 0; i < 8; i++) {
      int q = tid + i * 256;
      int row = q >> 4, kp = q & 15;
      int gr = bm + row;
      int grc = gr < M ? gr : M - 1;
      float2 v = *(const float2*)(X + (size_t)grc * K + kt + kp * 2);
      if (gr >= M) { v.x = 0.f; v.y = 0.f; }
      av[i] = v;
    }
  };
  auto storeA = [&]() {
#pragma unroll
    for (int i = 0; i < 8; i++) {
      int q = tid + i * 256;
      int row = q >> 4, kp = q & 15;
      As[row * LDA + kp * 2]     = (__bf16)av[i].x;
      As[row * LDA + kp * 2 + 1] = (__bf16)av[i].y;
    }
  };
  auto stageB = [&](int it) {
    float* dst = (it & 1) ? Bs1 : Bs0;
    int kt = it * 32;
#ifdef HAVE_TDM
    if (wave == 0) {
      unsigned lds_byte = (unsigned)((char*)dst - (char*)smem);
      tdm_load_tile(W + (size_t)kt * N + bn, lds_byte,
                    (unsigned)(N - bn), (unsigned)(K - kt), (unsigned)N);
    }
#else
#pragma unroll
    for (int i = 0; i < 16; i++) {
      int q = tid + i * 256;
      int k = q >> 7, n = q & 127;
      dst[k * LDB + n] = W[(size_t)(kt + k) * N + bn + n];
    }
#endif
  };

  loadA(0);
  stageB(0);

  for (int it = 0; it < nt; it++) {
    storeA();                         // tile 'it' A -> LDS (bf16)
    if (it + 1 < nt) loadA(it + 1);   // overlap next A global loads
    const float* Bcur = (it & 1) ? Bs1 : Bs0;
    if (it + 1 < nt) stageB(it + 1);  // overlap next B TDM copy
#ifdef HAVE_TDM
    if (wave == 0) {
      if (it + 1 < nt) __builtin_amdgcn_s_wait_tensorcnt(1);
      else             __builtin_amdgcn_s_wait_tensorcnt(0);
    }
#endif
    __syncthreads();

    v16bf afr[2];
#pragma unroll
    for (int m = 0; m < 2; m++) {
      const __bf16* base = &As[(wm * 32 + m * 16 + l16) * LDA + hf * 8];
      afr[m] = make_frag(*(const v8bf*)base, *(const v8bf*)(base + 16));
    }
#pragma unroll
    for (int j = 0; j < 4; j++) {
      int ncol = wn * 64 + j * 16 + l16;
      v16bf bfr;
#pragma unroll
      for (int e = 0; e < 8; e++) {
        bfr[e]     = (__bf16)Bcur[(hf * 8 + e) * LDB + ncol];
        bfr[8 + e] = (__bf16)Bcur[(16 + hf * 8 + e) * LDB + ncol];
      }
#pragma unroll
      for (int m = 0; m < 2; m++)
        acc[m][j] = __builtin_amdgcn_wmma_f32_16x16x32_bf16(
            false, afr[m], false, bfr, (short)0, acc[m][j], false, false);
    }
    __syncthreads();
  }

  // epilogue: C layout lane -> (row = v + 8*hf, col = l16)
#pragma unroll
  for (int m = 0; m < 2; m++) {
#pragma unroll
    for (int j = 0; j < 4; j++) {
      int gcol = bn + wn * 64 + j * 16 + l16;
      float bv = bias[gcol];
#pragma unroll
      for (int v = 0; v < 8; v++) {
        int grow = bm + wm * 32 + m * 16 + v + hf * 8;
        if (grow < M) {
          float r = acc[m][j][v] + bv;
          if (act == 1) r = gelu_tanh(r);
          Y[(size_t)grow * N + gcol] = r;
        }
      }
    }
  }
}

// =====================================================================
// Flash-attention, one wave per (16-query tile, head). WMMA for QK^T and PV.
// mode 0: cross attention (block-diag + NW=2 band), mode 1: frame-causal self.
// =====================================================================
__device__ inline bool mask_allow(int mode, int fq, int col) {
  if (mode == 1) { int fk = col / SEQL; return fk <= fq; }
  if (col < NZ)  { int fk = col / SEQL; return fk == fq; }
  int fk = (col - NZ) / SEQL;
  return (fk == fq - 1) || (fk == fq - 2);
}

__global__ __launch_bounds__(32) void attn_kernel(
    const float* __restrict__ Q, const float* __restrict__ Kb,
    const float* __restrict__ Vb, float* __restrict__ O,
    int Nq, int Nkv, int mode)
{
  __shared__ __bf16 Ps[16 * LDA];   // P tile 16x32 (bf16)
  __shared__ __bf16 Vt[64 * LDA];   // V tile transposed [feat][kv]

  const int lane = threadIdx.x;
  const int hf = lane >> 4, l16 = lane & 15;
  const int h = blockIdx.y;
  const int qbase = blockIdx.x * 16;

  // Q fragments, scaled by DH^-0.5 = 0.125
  v16bf qf[2];
  {
    int qrow = qbase + l16;
    bool ok = qrow < Nq;
    int qrc = ok ? qrow : Nq - 1;
#pragma unroll
    for (int kc = 0; kc < 2; kc++) {
      const float* p = Q + (size_t)qrc * DD + h * DHH + kc * 32 + hf * 8;
      qf[kc] = make_frag(cvt8sel(p, ok, 0.125f), cvt8sel(p + 16, ok, 0.125f));
    }
  }

  float mst[8], lst[8];
  v8f oacc[4];
#pragma unroll
  for (int v = 0; v < 8; v++) { mst[v] = -3.0e38f; lst[v] = 0.f; }
#pragma unroll
  for (int j = 0; j < 4; j++) oacc[j] = zero_v8f();

  const int nkt = (Nkv + 31) / 32;
  for (int kt = 0; kt < nkt; kt++) {
    const int kvb = kt * 32;
    v8f sacc[2];
    sacc[0] = zero_v8f(); sacc[1] = zero_v8f();

    // S = Q @ K^T : B fragments straight from row-major K
#pragma unroll
    for (int kc = 0; kc < 2; kc++) {
#pragma unroll
      for (int n = 0; n < 2; n++) {
        int krow = kvb + n * 16 + l16;
        bool ok = krow < Nkv;
        int krc = ok ? krow : Nkv - 1;
        const float* p = Kb + (size_t)krc * DD + h * DHH + kc * 32 + hf * 8;
        v16bf bfr = make_frag(cvt8sel(p, ok, 1.f), cvt8sel(p + 16, ok, 1.f));
        sacc[n] = __builtin_amdgcn_wmma_f32_16x16x32_bf16(
            false, qf[kc], false, bfr, (short)0, sacc[n], false, false);
      }
    }

    // online softmax per row (16-lane butterflies)
#pragma unroll
    for (int v = 0; v < 8; v++) {
      int qt = qbase + v + hf * 8;
      int fq = qt / SEQL;
      float s0 = sacc[0][v], s1 = sacc[1][v];
      int c0 = kvb + l16, c1 = kvb + 16 + l16;
      if (!(c0 < Nkv && mask_allow(mode, fq, c0))) s0 = -3.0e38f;
      if (!(c1 < Nkv && mask_allow(mode, fq, c1))) s1 = -3.0e38f;
      float rm = fmaxf(s0, s1);
#pragma unroll
      for (int mk = 1; mk < 16; mk <<= 1) rm = fmaxf(rm, __shfl_xor(rm, mk, 16));
      float nm = fmaxf(mst[v], rm);
      float p0 = (s0 <= -1.0e30f) ? 0.f : __expf(s0 - nm);
      float p1 = (s1 <= -1.0e30f) ? 0.f : __expf(s1 - nm);
      float rs = p0 + p1;
#pragma unroll
      for (int mk = 1; mk < 16; mk <<= 1) rs += __shfl_xor(rs, mk, 16);
      float sc = (mst[v] <= -1.0e30f) ? 0.f : __expf(mst[v] - nm);
      lst[v] = lst[v] * sc + rs;
      mst[v] = nm;
#pragma unroll
      for (int j = 0; j < 4; j++) oacc[j][v] *= sc;
      int prow = v + hf * 8;
      Ps[prow * LDA + l16]      = (__bf16)p0;
      Ps[prow * LDA + 16 + l16] = (__bf16)p1;
    }

    // stage V tile transposed (vectorized b128 loads, branchless)
    {
      int krow = kvb + lane;
      bool ok = krow < Nkv;
      int krc = ok ? krow : Nkv - 1;
      const float* vp = Vb + (size_t)krc * DD + h * DHH;
#pragma unroll
      for (int f4 = 0; f4 < 16; f4++) {
        float4 vv = *(const float4*)(vp + f4 * 4);
        if (!ok) { vv.x = 0.f; vv.y = 0.f; vv.z = 0.f; vv.w = 0.f; }
        Vt[(f4 * 4 + 0) * LDA + lane] = (__bf16)vv.x;
        Vt[(f4 * 4 + 1) * LDA + lane] = (__bf16)vv.y;
        Vt[(f4 * 4 + 2) * LDA + lane] = (__bf16)vv.z;
        Vt[(f4 * 4 + 3) * LDA + lane] = (__bf16)vv.w;
      }
    }
    __syncthreads();

    // O += P @ V
    {
      const __bf16* pb = &Ps[l16 * LDA + hf * 8];
      v16bf pf = make_frag(*(const v8bf*)pb, *(const v8bf*)(pb + 16));
#pragma unroll
      for (int j = 0; j < 4; j++) {
        const __bf16* vb = &Vt[(j * 16 + l16) * LDA + hf * 8];
        v16bf bfr = make_frag(*(const v8bf*)vb, *(const v8bf*)(vb + 16));
        oacc[j] = __builtin_amdgcn_wmma_f32_16x16x32_bf16(
            false, pf, false, bfr, (short)0, oacc[j], false, false);
      }
    }
    __syncthreads();
  }

#pragma unroll
  for (int j = 0; j < 4; j++) {
#pragma unroll
    for (int v = 0; v < 8; v++) {
      int qt = qbase + v + hf * 8;
      if (qt < Nq)
        O[(size_t)qt * DD + h * DHH + j * 16 + l16] =
            oacc[j][v] / fmaxf(lst[v], 1e-20f);
    }
  }
}

// =====================================================================
// small scalar kernels
// =====================================================================
__global__ void cond_kernel(const int* __restrict__ actions,
                            const float* __restrict__ ts,
                            const float* __restrict__ aemb,
                            const float* __restrict__ temb,
                            float* __restrict__ scond, float* __restrict__ sclean)
{
  int d = threadIdx.x;
  for (int t = 0; t < DUR; t++) {
    float a = aemb[actions[t] * DD + d];
    int idx = (int)(ts[t] * (float)TT);
    idx = idx < 0 ? 0 : (idx > TT - 1 ? TT - 1 : idx);
    float c = temb[(size_t)idx * DD + d] + a;
    scond[t * DD + d] = c / (1.f + __expf(-c));       // silu
    if (t < DUR - 1) {
      float cl = temb[d] + a;
      sclean[t * DD + d] = cl / (1.f + __expf(-cl));
    }
  }
}

__global__ void small_gemm(const float* __restrict__ S, const float* __restrict__ W,
                           const float* __restrict__ b, float* __restrict__ out,
                           int M, int N)
{
  int id = blockIdx.x * 256 + threadIdx.x;
  if (id >= M * N) return;
  int m = id / N, n = id - m * N;
  float acc = b[n];
  for (int k = 0; k < DD; k++) acc += S[m * DD + k] * W[(size_t)k * N + n];
  out[id] = acc;
}

__global__ void patchify_kernel(const float* __restrict__ src,
                                const float* __restrict__ pw, const float* __restrict__ pb,
                                const float* __restrict__ pe, const float* __restrict__ regs,
                                float* __restrict__ dst, int dur)
{
  int id = blockIdx.x * 256 + threadIdx.x;
  int total = dur * SEQL * DD;
  if (id >= total) return;
  int d = id & (DD - 1);
  int row = id >> 9;
  int t = row / SEQL, p = row - t * SEQL;
  float val;
  if (p < GRIDN) {
    int py = p >> 4, px = p & 15;
    float acc = pb[d];
#pragma unroll
    for (int c = 0; c < CCH; c++)
#pragma unroll
      for (int i = 0; i < PP; i++)
#pragma unroll
        for (int j = 0; j < PP; j++)
          acc += src[((size_t)(t * CCH + c) * HH + py * 2 + i) * WW + px * 2 + j] *
                 pw[d * 12 + c * 4 + i * 2 + j];
    val = acc + pe[p * DD + d];
  } else {
    val = regs[(p - GRIDN) * DD + d];
  }
  dst[(size_t)row * DD + d] = val;
}

__global__ __launch_bounds__(256) void adaln_kernel(const float* __restrict__ X,
                                                    const float* __restrict__ mod,
                                                    float* __restrict__ Y, int rows)
{
  __shared__ float s1[256], s2[256];
  int r = blockIdx.x;
  if (r >= rows) return;
  int tid = threadIdx.x;
  const float* x = X + (size_t)r * DD;
  float a = x[tid], b = x[tid + 256];
  s1[tid] = a + b;
  s2[tid] = a * a + b * b;
  __syncthreads();
  for (int st = 128; st > 0; st >>= 1) {
    if (tid < st) { s1[tid] += s1[tid + st]; s2[tid] += s2[tid + st]; }
    __syncthreads();
  }
  float mu = s1[0] * (1.f / 512.f);
  float var = s2[0] * (1.f / 512.f) - mu * mu;
  float inv = rsqrtf(var + 1e-5f);
  const float* md = mod + (r / SEQL) * (2 * DD);
  Y[(size_t)r * DD + tid]       = (a - mu) * inv * (1.f + md[DD + tid]) + md[tid];
  Y[(size_t)r * DD + tid + 256] = (b - mu) * inv * (1.f + md[DD + tid + 256]) + md[tid + 256];
}

__global__ void rope_kernel(float* __restrict__ buf, int rows, int split)
{
  int id = blockIdx.x * 256 + threadIdx.x;
  if (id >= rows * 256) return;
  int row = id >> 8;
  int r = id & 255;
  int h = r >> 5, i = r & 31;
  float* p = buf + (size_t)row * DD + h * DHH;
  int pos = (row < split) ? row / SEQL : (row - split) / SEQL;
  float freq = __powf(10000.f, -(float)(2 * i) / (float)DHH);
  float ang = (float)pos * freq;
  float cs = __cosf(ang), sn = __sinf(ang);
  float x1 = p[i], x2 = p[32 + i];
  p[i]      = x1 * cs - x2 * sn;
  p[32 + i] = x1 * sn + x2 * cs;
}

__global__ void gate_res_kernel(float* __restrict__ Y, const float* __restrict__ Xin,
                                const float* __restrict__ gate, int rows)
{
  int id = blockIdx.x * 256 + threadIdx.x;
  if (id >= rows * DD) return;
  int r = id >> 9, d = id & (DD - 1);
  Y[id] += Xin[id] * gate[(r / SEQL) * DD + d];
}

__global__ void mul_kernel(float* __restrict__ A, const float* __restrict__ B, int n)
{
  int id = blockIdx.x * 256 + threadIdx.x;
  if (id < n) A[id] *= B[id];
}

__global__ void unpatch_kernel(const float* __restrict__ zrf,
                               const float* __restrict__ uw, const float* __restrict__ ub,
                               float* __restrict__ out)
{
  int id = blockIdx.x * 256 + threadIdx.x;
  if (id >= DUR * CCH * HH * WW) return;
  int w = id & 31;
  int t1 = id >> 5;
  int h = t1 & 31;
  int t2 = t1 >> 5;
  int c = t2 % CCH;
  int t = t2 / CCH;
  int py = h >> 1, i = h & 1, px = w >> 1, j = w & 1;
  int row = t * SEQL + py * 16 + px;
  int col = c * 4 + i * 2 + j;
  float acc = ub[col];
  for (int k = 0; k < DD; k++) acc += zrf[(size_t)row * DD + k] * uw[k * 12 + col];
  out[id] = acc;
}

// =====================================================================
// host orchestration
// =====================================================================
static inline int cdiv(int a, int b) { return (a + b - 1) / b; }

extern "C" void kernel_launch(void* const* d_in, const int* in_sizes, int n_in,
                              void* d_out, int out_size, void* d_ws, size_t ws_size,
                              hipStream_t stream)
{
  (void)in_sizes; (void)n_in; (void)out_size; (void)ws_size;

  const float* z       = (const float*)d_in[0];
  const float* frames  = (const float*)d_in[1];
  const int*   actions = (const int*)d_in[2];
  const float* ts      = (const float*)d_in[3];
  const float* aemb    = (const float*)d_in[4];
  const float* temb    = (const float*)d_in[5];
  const float* regs    = (const float*)d_in[6];
  const float* pe      = (const float*)d_in[7];
  const float* pw      = (const float*)d_in[8];
  const float* pb      = (const float*)d_in[9];
  const float* uw      = (const float*)d_in[10];
  const float* ub      = (const float*)d_in[11];
  const float* a1w = (const float*)d_in[12]; const float* a1b = (const float*)d_in[13];
  const float* a2w = (const float*)d_in[14]; const float* a2b = (const float*)d_in[15];
  const float* g1w = (const float*)d_in[16]; const float* g1b = (const float*)d_in[17];
  const float* g2w = (const float*)d_in[18]; const float* g2b = (const float*)d_in[19];
  const float* caqw = (const float*)d_in[20]; const float* caqb = (const float*)d_in[21];
  const float* cakw = (const float*)d_in[22]; const float* cakb = (const float*)d_in[23];
  const float* cavw = (const float*)d_in[24]; const float* cavb = (const float*)d_in[25];
  const float* caow = (const float*)d_in[26]; const float* caob = (const float*)d_in[27];
  const float* saqw = (const float*)d_in[28]; const float* saqb = (const float*)d_in[29];
  const float* sakw = (const float*)d_in[30]; const float* sakb = (const float*)d_in[31];
  const float* savw = (const float*)d_in[32]; const float* savb = (const float*)d_in[33];
  const float* saow = (const float*)d_in[34]; const float* saob = (const float*)d_in[35];
  const float* gwg = (const float*)d_in[36]; const float* gbg = (const float*)d_in[37];
  const float* gwu = (const float*)d_in[38]; const float* gbu = (const float*)d_in[39];
  const float* gwo = (const float*)d_in[40]; const float* gbo = (const float*)d_in[41];

  float* ws  = (float*)d_ws;
  float* ZRF = ws;
  float* XAF = ZRF + (size_t)NZ * DD;
  float* N1  = XAF + (size_t)NX * DD;
  float* Qb  = N1  + (size_t)NKV * DD;
  float* Kb  = Qb  + (size_t)NZ * DD;
  float* Vb  = Kb  + (size_t)NKV * DD;
  float* ATT = Vb  + (size_t)NKV * DD;
  float* PRJ = ATT + (size_t)NKV * DD;
  float* G1  = PRJ + (size_t)NKV * DD;
  float* U1  = G1  + (size_t)NKV * 2048;
  float* SC  = U1  + (size_t)NKV * 2048;
  float* SCL = SC  + 6 * DD;
  float* MZ1 = SCL + 5 * DD;
  float* MX1 = MZ1 + 6 * 1024;
  float* GZ1 = MX1 + 5 * 1024;
  float* GX1 = GZ1 + 6 * DD;
  float* MZ2 = GX1 + 5 * DD;
  float* MX2 = MZ2 + 6 * 1024;
  float* GZ2 = MX2 + 5 * 1024;
  float* GX2 = GZ2 + 6 * DD;

  auto gemm = [&](const float* X, const float* Wm, const float* bi, float* Y,
                  int M, int N, int K, int act) {
    dim3 g(cdiv(M, 128), N / 128);
    gemm_bf16<<<g, 256, 0, stream>>>(X, Wm, bi, Y, M, N, K, act);
  };
  auto sgemm = [&](const float* S, const float* Wm, const float* bi, float* Y,
                   int M, int N) {
    small_gemm<<<cdiv(M * N, 256), 256, 0, stream>>>(S, Wm, bi, Y, M, N);
  };

  cond_kernel<<<1, 512, 0, stream>>>(actions, ts, aemb, temb, SC, SCL);
  patchify_kernel<<<cdiv(NZ * DD, 256), 256, 0, stream>>>(z, pw, pb, pe, regs, ZRF, DUR);
  patchify_kernel<<<cdiv(NX * DD, 256), 256, 0, stream>>>(frames, pw, pb, pe, regs, XAF, DUR - 1);

  const int BIG = 1 << 30;
  for (int bk = 0; bk < NBLK; bk++) {
    const size_t wDD = (size_t)DD * DD;
    const size_t w2D = (size_t)DD * 2 * DD;
    const size_t wE  = (size_t)DD * EXPF * DD;

    sgemm(SC,  a1w + bk * w2D, a1b + bk * 2 * DD, MZ1, DUR, 2 * DD);
    sgemm(SCL, a1w + bk * w2D, a1b + bk * 2 * DD, MX1, DUR - 1, 2 * DD);
    sgemm(SC,  g1w + bk * wDD, g1b + bk * DD, GZ1, DUR, DD);
    sgemm(SCL, g1w + bk * wDD, g1b + bk * DD, GX1, DUR - 1, DD);
    sgemm(SC,  a2w + bk * w2D, a2b + bk * 2 * DD, MZ2, DUR, 2 * DD);
    sgemm(SCL, a2w + bk * w2D, a2b + bk * 2 * DD, MX2, DUR - 1, 2 * DD);
    sgemm(SC,  g2w + bk * wDD, g2b + bk * DD, GZ2, DUR, DD);
    sgemm(SCL, g2w + bk * wDD, g2b + bk * DD, GX2, DUR - 1, DD);

    adaln_kernel<<<NZ, 256, 0, stream>>>(ZRF, MZ1, N1, NZ);
    adaln_kernel<<<NX, 256, 0, stream>>>(XAF, MX1, N1 + (size_t)NZ * DD, NX);

    // cross attention
    gemm(N1, caqw + bk * wDD, caqb + bk * DD, Qb, NZ,  DD, DD, 0);
    gemm(N1, cakw + bk * wDD, cakb + bk * DD, Kb, NKV, DD, DD, 0);
    gemm(N1, cavw + bk * wDD, cavb + bk * DD, Vb, NKV, DD, DD, 0);
    rope_kernel<<<cdiv(NZ * 256, 256), 256, 0, stream>>>(Qb, NZ, BIG);
    rope_kernel<<<cdiv(NKV * 256, 256), 256, 0, stream>>>(Kb, NKV, NZ);
    attn_kernel<<<dim3(cdiv(NZ, 16), NHH), 32, 0, stream>>>(Qb, Kb, Vb, ATT, NZ, NKV, 0);

    // self attention
    const float* XN = N1 + (size_t)NZ * DD;
    gemm(XN, saqw + bk * wDD, saqb + bk * DD, Qb, NX, DD, DD, 0);
    gemm(XN, sakw + bk * wDD, sakb + bk * DD, Kb, NX, DD, DD, 0);
    gemm(XN, savw + bk * wDD, savb + bk * DD, Vb, NX, DD, DD, 0);
    rope_kernel<<<cdiv(NX * 256, 256), 256, 0, stream>>>(Qb, NX, BIG);
    rope_kernel<<<cdiv(NX * 256, 256), 256, 0, stream>>>(Kb, NX, BIG);
    attn_kernel<<<dim3(cdiv(NX, 16), NHH), 32, 0, stream>>>(
        Qb, Kb, Vb, ATT + (size_t)NZ * DD, NX, NX, 1);

    // output projections + gate + residual
    gemm(ATT, caow + bk * wDD, caob + bk * DD, PRJ, NZ, DD, DD, 0);
    gemm(ATT + (size_t)NZ * DD, saow + bk * wDD, saob + bk * DD,
         PRJ + (size_t)NZ * DD, NX, DD, DD, 0);
    gate_res_kernel<<<cdiv(NZ * DD, 256), 256, 0, stream>>>(ZRF, PRJ, GZ1, NZ);
    gate_res_kernel<<<cdiv(NX * DD, 256), 256, 0, stream>>>(
        XAF, PRJ + (size_t)NZ * DD, GX1, NX);

    // AdaLN 2 + GEGLU (combined z/x rows, shared weights)
    adaln_kernel<<<NZ, 256, 0, stream>>>(ZRF, MZ2, N1, NZ);
    adaln_kernel<<<NX, 256, 0, stream>>>(XAF, MX2, N1 + (size_t)NZ * DD, NX);
    gemm(N1, gwg + bk * wE, gbg + bk * EXPF * DD, G1, NKV, EXPF * DD, DD, 1);
    gemm(N1, gwu + bk * wE, gbu + bk * EXPF * DD, U1, NKV, EXPF * DD, DD, 0);
    mul_kernel<<<cdiv(NKV * EXPF * DD, 256), 256, 0, stream>>>(G1, U1, NKV * EXPF * DD);
    gemm(G1, gwo + bk * wE, gbo + bk * DD, PRJ, NKV, DD, EXPF * DD, 0);
    gate_res_kernel<<<cdiv(NZ * DD, 256), 256, 0, stream>>>(ZRF, PRJ, GZ2, NZ);
    gate_res_kernel<<<cdiv(NX * DD, 256), 256, 0, stream>>>(
        XAF, PRJ + (size_t)NZ * DD, GX2, NX);
  }

  unpatch_kernel<<<cdiv(DUR * CCH * HH * WW, 256), 256, 0, stream>>>(
      ZRF, uw, ub, (float*)d_out);
}
